// DLRM_20779051778714
// MI455X (gfx1250) — matrix operand order repaired and checked by reading, not compiled, not run on hardware
//
#include <hip/hip_runtime.h>
#include <hip/hip_bf16.h>
#include <math.h>

// ---------------------------------------------------------------------------
// DLRM forward, fused into 5 kernel types, bf16 WMMA (gfx1250) everywhere a
// matmul appears. fp32 accumulate, bf16 activation chaining, fp32 output.
// GEMM waves own a 32(M)x64(N) strip: 2 A-tiles x 4 B-tiles -> 8 WMMAs/K-chunk.
// ---------------------------------------------------------------------------

#define NUM_SPARSE 26
#define VOCAB      100000
#define EMBD       64
#define BATCH      2048
#define DENSE_IN   13
#define NUM_FEA    27          // 1 + NUM_SPARSE
#define R_COLS     416         // 64 + 351 lower-tri + 1 zero pad (multiple of 32)

typedef __bf16 bf16;
typedef bf16  v8bf  __attribute__((ext_vector_type(8)));
typedef bf16  v16bf __attribute__((ext_vector_type(16)));
typedef float v8f   __attribute__((ext_vector_type(8)));

union V16U { v16bf v; v8bf h[2]; };

// A-tile (16x32, M x K): lane L holds row (L&15); elements 0..7 = K kb+8*half+e,
// elements 8..15 = K kb+16+8*half+e  (per CDNA5 ISA 16-bit A layout).
__device__ __forceinline__ v16bf load_a_tile(const bf16* __restrict__ row, int kb, int half) {
  V16U u;
  u.h[0] = *(const v8bf*)(row + kb +      half * 8);
  u.h[1] = *(const v8bf*)(row + kb + 16 + half * 8);
  return u.v;
}

// B-tile (32x16, K x N): lane L holds column (L&15); elements 0..7 = K kb+8*half+e,
// elements 8..15 = K kb+16+8*half+e (same K order as the A-tile).
// `col` points at a contiguous K-major column (weights stored transposed N x K).
__device__ __forceinline__ v16bf load_b_tile(const bf16* __restrict__ col, int kb, int half) {
  V16U u;
  const bf16* p = col + kb + half * 8;
  u.h[0] = *(const v8bf*)(p);
  u.h[1] = *(const v8bf*)(p + 16);
  return u.v;
}

__device__ __forceinline__ v16bf zero_v16() {
  V16U u;
#pragma unroll
  for (int i = 0; i < 8; ++i) { u.h[0][i] = (bf16)0.0f; u.h[1][i] = (bf16)0.0f; }
  return u.v;
}

__device__ __forceinline__ v8f wmma_bf16(v16bf a, v16bf b, v8f c) {
  return __builtin_amdgcn_wmma_f32_16x16x32_bf16(false, a, false, b, (short)0, c,
                                                 false, false);
}

// C/D layout: VGPR v, lanes 0-15 -> (M=v, N=lane); lanes 16-31 -> (M=v+8, N=lane-16)
__device__ __forceinline__ void store_tile(v8f c, bf16* __restrict__ out,
                                           const float* __restrict__ bias,
                                           int N, int m0, int n0, int lane, int relu) {
  const int n     = n0 + (lane & 15);
  const int mbase = m0 + ((lane >> 4) << 3);
  const float b   = bias[n];
#pragma unroll
  for (int v = 0; v < 8; ++v) {
    float x = c[v] + b;
    if (relu) x = fmaxf(x, 0.0f);
    out[(size_t)(mbase + v) * N + n] = (bf16)x;
  }
}

// ---------------------------------------------------------------------------
// bf16 GEMM: out[M,N] = act(A[M,K] @ W[K,N] + bias), W given as Wt[N,K].
// 8 waves / block; each wave computes a 32(M) x 64(N) strip: per K-chunk two
// A-tiles are amortized over four shared B-tiles (8 WMMAs / 6KB loads).
// ---------------------------------------------------------------------------
__global__ void __launch_bounds__(256)
gemm_bias_relu(const bf16* __restrict__ A, const bf16* __restrict__ Wt,
               const float* __restrict__ bias, bf16* __restrict__ out,
               int M, int N, int K, int relu)
{
  const int lane   = threadIdx.x & 31;
  const int gw     = blockIdx.x * 8 + (threadIdx.x >> 5);
  const int mtiles = M >> 5;                 // 32-row strips
  const int mt     = gw % mtiles;
  const int nt     = gw / mtiles;
  const int n0     = nt * 64;
  if (n0 >= N) return;

  const int half = lane >> 4;
  const int l15  = lane & 15;
  const int m0   = mt * 32;

  const bf16* arow0 = A + (size_t)(m0 +      l15) * K;
  const bf16* arow1 = A + (size_t)(m0 + 16 + l15) * K;
  const bf16* col0  = Wt + (size_t)(n0 +  0 + l15) * K;
  const bf16* col1  = Wt + (size_t)(n0 + 16 + l15) * K;
  const bf16* col2  = Wt + (size_t)(n0 + 32 + l15) * K;
  const bf16* col3  = Wt + (size_t)(n0 + 48 + l15) * K;

  v8f c00 = {}, c01 = {}, c02 = {}, c03 = {};
  v8f c10 = {}, c11 = {}, c12 = {}, c13 = {};

  for (int kb = 0; kb < K; kb += 32) {
    if (kb + 128 < K) {                      // stage next chunks into caches
      __builtin_prefetch(arow0 + kb + 128, 0, 3);
      __builtin_prefetch(arow1 + kb + 128, 0, 3);
    }
    v16bf a0 = load_a_tile(arow0, kb, half);
    v16bf a1 = load_a_tile(arow1, kb, half);
    v16bf b0 = load_b_tile(col0, kb, half);
    v16bf b1 = load_b_tile(col1, kb, half);
    v16bf b2 = load_b_tile(col2, kb, half);
    v16bf b3 = load_b_tile(col3, kb, half);
    c00 = wmma_bf16(a0, b0, c00);
    c01 = wmma_bf16(a0, b1, c01);
    c02 = wmma_bf16(a0, b2, c02);
    c03 = wmma_bf16(a0, b3, c03);
    c10 = wmma_bf16(a1, b0, c10);
    c11 = wmma_bf16(a1, b1, c11);
    c12 = wmma_bf16(a1, b2, c12);
    c13 = wmma_bf16(a1, b3, c13);
  }

  store_tile(c00, out, bias, N, m0,      n0 +  0, lane, relu);
  store_tile(c01, out, bias, N, m0,      n0 + 16, lane, relu);
  store_tile(c02, out, bias, N, m0,      n0 + 32, lane, relu);
  store_tile(c03, out, bias, N, m0,      n0 + 48, lane, relu);
  store_tile(c10, out, bias, N, m0 + 16, n0 +  0, lane, relu);
  store_tile(c11, out, bias, N, m0 + 16, n0 + 16, lane, relu);
  store_tile(c12, out, bias, N, m0 + 16, n0 + 32, lane, relu);
  store_tile(c13, out, bias, N, m0 + 16, n0 + 48, lane, relu);
}

// ---------------------------------------------------------------------------
// Embedding gather: one wave per (sample,feature) row; float2/lane -> bf16.
// HBM-bound: 2048*26 random 256B rows out of 665MB of tables.
// ---------------------------------------------------------------------------
__global__ void __launch_bounds__(256)
gather_emb(const int* __restrict__ cat, const float* __restrict__ tables,
           bf16* __restrict__ emb)
{
  const int t    = blockIdx.x * blockDim.x + threadIdx.x;
  const int lane = t & 31;
  const int row  = t >> 5;                 // row = s*26 + f
  const int f    = row % NUM_SPARSE;
  const int idx  = cat[row];
  const float2 x = *(const float2*)(tables + ((size_t)f * VOCAB + idx) * EMBD + lane * 2);
  bf16* d = emb + (size_t)row * EMBD + lane * 2;
  d[0] = (bf16)x.x;
  d[1] = (bf16)x.y;
}

// ---------------------------------------------------------------------------
// Interaction: one wave per sample. T = [bot; emb] is 27x64 (padded to 32).
// Z = T T^T via WMMA: B-tile columns of T^T are rows of T, so we load T rows
// in both A- and B-layout and compute C00, C10, C11 (C01 is symmetric).
// Emits R = [bot(64) | tril(Z) (351) | 0] in bf16, width 416.
// ---------------------------------------------------------------------------
__device__ __forceinline__ void emit_tri(v8f c, int ti, int tj, int lane,
                                         bf16* __restrict__ rrow) {
  const int j  = tj * 16 + (lane & 15);
  const int ib = ti * 16 + ((lane >> 4) << 3);
#pragma unroll
  for (int v = 0; v < 8; ++v) {
    const int i = ib + v;
    if (i < NUM_FEA && j < i)
      rrow[64 + (i * (i - 1)) / 2 + j] = (bf16)c[v];
  }
}

__global__ void __launch_bounds__(256)
interact_kernel(const bf16* __restrict__ bot, const bf16* __restrict__ emb,
                bf16* __restrict__ R)
{
  const int lane = threadIdx.x & 31;
  const int s    = blockIdx.x * 8 + (threadIdx.x >> 5);
  const int half = lane >> 4;
  const int l15  = lane & 15;

  const bf16* bots = bot + (size_t)s * EMBD;
  const bf16* embs = emb + (size_t)s * NUM_SPARSE * EMBD;

  // T row pointers for this lane: rows l15 (block0) and 16+l15 (block1)
  const bf16* p0 = (l15 == 0) ? bots : (embs + (size_t)(l15 - 1) * EMBD);
  const int   r1 = 16 + l15;
  const bool  ok1 = (r1 < NUM_FEA);
  const bf16* p1 = embs + (size_t)(r1 - 1) * EMBD;

  v16bf a0k0 = load_a_tile(p0, 0, half);
  v16bf a0k1 = load_a_tile(p0, 32, half);
  v16bf b0k0 = load_b_tile(p0, 0, half);
  v16bf b0k1 = load_b_tile(p0, 32, half);

  v16bf a1k0, a1k1, b1k0, b1k1;
  if (ok1) {
    a1k0 = load_a_tile(p1, 0, half);
    a1k1 = load_a_tile(p1, 32, half);
    b1k0 = load_b_tile(p1, 0, half);
    b1k1 = load_b_tile(p1, 32, half);
  } else {
    a1k0 = zero_v16(); a1k1 = zero_v16();
    b1k0 = zero_v16(); b1k1 = zero_v16();
  }

  v8f c00 = {}, c10 = {}, c11 = {};
  c00 = wmma_bf16(a0k0, b0k0, c00);  c00 = wmma_bf16(a0k1, b0k1, c00);
  c10 = wmma_bf16(a1k0, b0k0, c10);  c10 = wmma_bf16(a1k1, b0k1, c10);
  c11 = wmma_bf16(a1k0, b1k0, c11);  c11 = wmma_bf16(a1k1, b1k1, c11);

  bf16* rrow = R + (size_t)s * R_COLS;
  for (int c = lane; c < EMBD; c += 32) rrow[c] = bots[c];
  if (lane == 0) rrow[R_COLS - 1] = (bf16)0.0f;

  emit_tri(c00, 0, 0, lane, rrow);
  emit_tri(c10, 1, 0, lane, rrow);
  emit_tri(c11, 1, 1, lane, rrow);
}

// ---------------------------------------------------------------------------
// Final 256 -> 1 layer + sigmoid: one wave per sample, fp32 VALU dot +
// wave32 shuffle reduction. N=1 is a bad fit for WMMA; VALU is free here.
// ---------------------------------------------------------------------------
__global__ void __launch_bounds__(256)
final_layer(const bf16* __restrict__ act, const float* __restrict__ w,
            const float* __restrict__ b, float* __restrict__ out)
{
  const int lane = threadIdx.x & 31;
  const int s    = blockIdx.x * 8 + (threadIdx.x >> 5);
  const bf16* a  = act + (size_t)s * 256;
  float acc = 0.0f;
#pragma unroll
  for (int t = 0; t < 8; ++t) {
    const int k = t * 32 + lane;
    acc += (float)a[k] * w[k];
  }
#pragma unroll
  for (int off = 16; off > 0; off >>= 1)
    acc += __shfl_down(acc, off, 32);
  if (lane == 0) out[s] = 1.0f / (1.0f + expf(-(acc + b[0])));
}

// ---------------------------------------------------------------------------
// Conversion helpers: fp32 W[K,N] -> bf16 Wt[N,Kpad] (zero-padded K), and
// fp32 rows -> bf16 zero-padded rows (dense input 13 -> 32).
// ---------------------------------------------------------------------------
__global__ void __launch_bounds__(256)
transpose_convert(const float* __restrict__ W, bf16* __restrict__ Wt,
                  int K, int N, int Kpad)
{
  const int t = blockIdx.x * blockDim.x + threadIdx.x;
  if (t >= N * Kpad) return;
  const int n = t / Kpad;
  const int k = t - n * Kpad;
  Wt[t] = (bf16)((k < K) ? W[(size_t)k * N + n] : 0.0f);
}

__global__ void __launch_bounds__(256)
pad_convert_rows(const float* __restrict__ src, bf16* __restrict__ dst,
                 int rows, int cols, int colsPad)
{
  const int t = blockIdx.x * blockDim.x + threadIdx.x;
  if (t >= rows * colsPad) return;
  const int r = t / colsPad;
  const int c = t - r * colsPad;
  dst[t] = (bf16)((c < cols) ? src[(size_t)r * cols + c] : 0.0f);
}

// ---------------------------------------------------------------------------
// Host-side orchestration
// ---------------------------------------------------------------------------
static inline int cdiv(int a, int b) { return (a + b - 1) / b; }

extern "C" void kernel_launch(void* const* d_in, const int* in_sizes, int n_in,
                              void* d_out, int out_size, void* d_ws, size_t ws_size,
                              hipStream_t stream)
{
  const float* dense = (const float*)d_in[0];
  const int*   cat   = (const int*)  d_in[1];
  const float* tabs  = (const float*)d_in[2];
  const float* bw0 = (const float*)d_in[3];  const float* bb0 = (const float*)d_in[4];
  const float* bw1 = (const float*)d_in[5];  const float* bb1 = (const float*)d_in[6];
  const float* bw2 = (const float*)d_in[7];  const float* bb2 = (const float*)d_in[8];
  const float* tw0 = (const float*)d_in[9];  const float* tb0 = (const float*)d_in[10];
  const float* tw1 = (const float*)d_in[11]; const float* tb1 = (const float*)d_in[12];
  const float* tw2 = (const float*)d_in[13]; const float* tb2 = (const float*)d_in[14];
  const float* tw3 = (const float*)d_in[15]; const float* tb3 = (const float*)d_in[16];
  float* out = (float*)d_out;

  // workspace layout (bf16 elements; every region 256B aligned by construction)
  char* ws = (char*)d_ws;
  size_t off = 0;
  auto alloc = [&](size_t elems) { bf16* p = (bf16*)(ws + off); off += elems * sizeof(bf16); return p; };
  bf16* dpad  = alloc((size_t)BATCH * 32);            // padded dense
  bf16* wtb0  = alloc((size_t)512 * 32);
  bf16* wtb1  = alloc((size_t)256 * 512);
  bf16* wtb2  = alloc((size_t)64 * 256);
  bf16* wtt0  = alloc((size_t)1024 * R_COLS);
  bf16* wtt1  = alloc((size_t)512 * 1024);
  bf16* wtt2  = alloc((size_t)256 * 512);
  bf16* h0    = alloc((size_t)BATCH * 512);
  bf16* h1    = alloc((size_t)BATCH * 256);
  bf16* bot   = alloc((size_t)BATCH * 64);
  bf16* emb   = alloc((size_t)BATCH * NUM_SPARSE * EMBD);
  bf16* R     = alloc((size_t)BATCH * R_COLS);
  bf16* t0    = alloc((size_t)BATCH * 1024);
  bf16* t1    = alloc((size_t)BATCH * 512);
  bf16* t2    = alloc((size_t)BATCH * 256);
  (void)ws_size; (void)n_in; (void)in_sizes; (void)out_size;

  // --- stage 0: weight / input conversion -------------------------------
  pad_convert_rows<<<cdiv(BATCH * 32, 256), 256, 0, stream>>>(dense, dpad, BATCH, DENSE_IN, 32);
  transpose_convert<<<cdiv(512 * 32, 256),      256, 0, stream>>>(bw0, wtb0, DENSE_IN, 512, 32);
  transpose_convert<<<cdiv(256 * 512, 256),     256, 0, stream>>>(bw1, wtb1, 512, 256, 512);
  transpose_convert<<<cdiv(64 * 256, 256),      256, 0, stream>>>(bw2, wtb2, 256, 64, 256);
  transpose_convert<<<cdiv(1024 * R_COLS, 256), 256, 0, stream>>>(tw0, wtt0, 415, 1024, R_COLS);
  transpose_convert<<<cdiv(512 * 1024, 256),    256, 0, stream>>>(tw1, wtt1, 1024, 512, 1024);
  transpose_convert<<<cdiv(256 * 512, 256),     256, 0, stream>>>(tw2, wtt2, 512, 256, 512);

  // --- stage 1: embedding gather (HBM-bound) ----------------------------
  gather_emb<<<(BATCH * NUM_SPARSE * 32) / 256, 256, 0, stream>>>(cat, tabs, emb);

  // --- stage 2: bottom MLP (WMMA) ---------------------------------------
  const int mt32 = BATCH / 32;  // 64 m-strips
  gemm_bias_relu<<<mt32 * (512 / 64) / 8,  256, 0, stream>>>(dpad, wtb0, bb0, h0,  BATCH, 512,  32,  1);
  gemm_bias_relu<<<mt32 * (256 / 64) / 8,  256, 0, stream>>>(h0,   wtb1, bb1, h1,  BATCH, 256,  512, 1);
  gemm_bias_relu<<<mt32 * (64  / 64) / 8,  256, 0, stream>>>(h1,   wtb2, bb2, bot, BATCH, 64,   256, 1);

  // --- stage 3: interaction (WMMA T T^T per sample) ---------------------
  interact_kernel<<<BATCH / 8, 256, 0, stream>>>(bot, emb, R);

  // --- stage 4: top MLP (WMMA) ------------------------------------------
  gemm_bias_relu<<<mt32 * (1024 / 64) / 8, 256, 0, stream>>>(R,  wtt0, tb0, t0, BATCH, 1024, R_COLS, 1);
  gemm_bias_relu<<<mt32 * (512 / 64) / 8,  256, 0, stream>>>(t0, wtt1, tb1, t1, BATCH, 512,  1024,   1);
  gemm_bias_relu<<<mt32 * (256 / 64) / 8,  256, 0, stream>>>(t1, wtt2, tb2, t2, BATCH, 256,  512,    1);

  // --- stage 5: final dot + sigmoid -------------------------------------
  final_layer<<<BATCH / 8, 256, 0, stream>>>(t2, tw3, tb3, out);
}
